// AttentionBlock_56642028700069
// MI455X (gfx1250) — matrix operand order repaired and checked by reading, not compiled, not run on hardware
//
#include <hip/hip_runtime.h>

// ---------------------------------------------------------------------------
// Flash attention forward, MI455X (gfx1250), wave32, bf16 WMMA, fp32 accum.
// B=2, H=12, S=2048, D=64, scale = 0.125.
// Per workgroup: 8 waves x 16 query rows = 128 rows. KV tile Bc = 64.
// Row-sum computed on the matrix pipe (P x ones), row-max via pipelined
// step-major shuffle reduction. 18 WMMAs per KV iteration.
// ---------------------------------------------------------------------------

typedef __attribute__((ext_vector_type(16))) __bf16 v16bf;
typedef __attribute__((ext_vector_type(8)))  float  v8f;

namespace {
constexpr int kB = 2;
constexpr int kH = 12;
constexpr int kS = 2048;
constexpr int kD = 64;
constexpr int kBR = 128;   // query rows per workgroup (8 waves x 16 rows)
constexpr int kBC = 64;    // kv columns per iteration
constexpr float kScale = 0.125f;
}

union FragU {
  uint4 u[2];
  v16bf v;
};

__device__ __forceinline__ unsigned short f2bf(float x) {
  __bf16 b = (__bf16)x;                 // hardware f32->bf16 conversion
  unsigned short u;
  __builtin_memcpy(&u, &b, 2);
  return u;
}

__device__ __forceinline__ unsigned int pk2bf(float lo, float hi) {
  return (unsigned int)f2bf(lo) | ((unsigned int)f2bf(hi) << 16);
}

__device__ __forceinline__ FragU ld_frag(const unsigned short* src) {
  FragU f;
  f.u[0] = ((const uint4*)src)[0];
  f.u[1] = ((const uint4*)src)[1];
  return f;
}

__global__ __launch_bounds__(256)
void fa_fwd_gfx1250(const float* __restrict__ q,
                    const float* __restrict__ k,
                    const float* __restrict__ v,
                    float* __restrict__ out) {
  __shared__ alignas(16) unsigned short ldsK[kBC][kD];        // [n][d]   8 KB
  __shared__ alignas(16) unsigned short ldsVT[kD][kBC];       // [d][n]   8 KB
  __shared__ alignas(16) unsigned short ldsP[8][16][kBC];     // per-wave 16 KB

  const int tid  = threadIdx.x;
  const int w    = tid >> 5;       // wave id 0..7
  const int lane = tid & 31;
  const int col  = lane & 15;
  const int half = lane >> 4;

  const int bh = blockIdx.y;                      // 0..23
  const int m0 = blockIdx.x * kBR + w * 16;       // first query row of wave
  const size_t base = (size_t)bh * kS * kD;

  // ---- Q fragment: 16x64 rows m0..m0+15, pre-scaled, bf16 A layout ----
  // A 16x32, contiguous k order used here (B matches): lane holds row M=col; elem i -> K = i + 16*half
  v16bf qfrag[2];
  {
    const float* qrow = q + base + (size_t)(m0 + col) * kD;
    #pragma unroll
    for (int c = 0; c < 2; ++c) {
      unsigned short tmp[16];
      #pragma unroll
      for (int i = 0; i < 16; ++i) {
        int d = c * 32 + 16 * half + i;
        tmp[i] = f2bf(qrow[d] * kScale);
      }
      FragU f;
      __builtin_memcpy(&f, tmp, 32);
      qfrag[c] = f.v;
    }
  }

  // ones B-fragment (bf16 1.0 everywhere) for row-sum on the matrix pipe
  FragU ones;
  ones.u[0] = make_uint4(0x3F803F80u, 0x3F803F80u, 0x3F803F80u, 0x3F803F80u);
  ones.u[1] = ones.u[0];

  // ---- accumulators ----
  v8f o[4];
  float m_r[8], l_r[8];
  #pragma unroll
  for (int t = 0; t < 4; ++t)
    #pragma unroll
    for (int r = 0; r < 8; ++r) o[t][r] = 0.0f;
  #pragma unroll
  for (int r = 0; r < 8; ++r) { m_r[r] = -1e30f; l_r[r] = 0.0f; }

  // ---- main loop over KV tiles of 64 columns ----
  #pragma unroll 1
  for (int j0 = 0; j0 < kS; j0 += kBC) {
    __syncthreads();  // previous tile fully consumed

    // Cooperative load: each thread converts 16 K floats + 16 V floats.
    {
      const int n  = tid >> 2;           // 0..63
      const int d0 = (tid & 3) * 16;     // 0,16,32,48
      const float* kp = k + base + (size_t)(j0 + n) * kD + d0;
      const float* vp = v + base + (size_t)(j0 + n) * kD + d0;
      float4 kq0 = ((const float4*)kp)[0];
      float4 kq1 = ((const float4*)kp)[1];
      float4 kq2 = ((const float4*)kp)[2];
      float4 kq3 = ((const float4*)kp)[3];
      uint4* kd = (uint4*)&ldsK[n][d0];
      kd[0] = make_uint4(pk2bf(kq0.x, kq0.y), pk2bf(kq0.z, kq0.w),
                         pk2bf(kq1.x, kq1.y), pk2bf(kq1.z, kq1.w));
      kd[1] = make_uint4(pk2bf(kq2.x, kq2.y), pk2bf(kq2.z, kq2.w),
                         pk2bf(kq3.x, kq3.y), pk2bf(kq3.z, kq3.w));
      float4 vq0 = ((const float4*)vp)[0];
      float4 vq1 = ((const float4*)vp)[1];
      float4 vq2 = ((const float4*)vp)[2];
      float4 vq3 = ((const float4*)vp)[3];
      float vv[16] = {vq0.x, vq0.y, vq0.z, vq0.w, vq1.x, vq1.y, vq1.z, vq1.w,
                      vq2.x, vq2.y, vq2.z, vq2.w, vq3.x, vq3.y, vq3.z, vq3.w};
      #pragma unroll
      for (int i = 0; i < 16; ++i) ldsVT[d0 + i][n] = f2bf(vv[i]);
      if (j0 + kBC < kS) {                          // CDNA5 global_prefetch_b8
        __builtin_prefetch(kp + kBC * kD, 0, 1);
        __builtin_prefetch(vp + kBC * kD, 0, 1);
      }
    }
    __syncthreads();

    // ---- scores: four 16x16 tiles (N halves), K-dim 64 = 2 WMMAs each ----
    v8f s[4];
    #pragma unroll
    for (int nh = 0; nh < 4; ++nh) {
      v8f acc;
      #pragma unroll
      for (int r = 0; r < 8; ++r) acc[r] = 0.0f;
      FragU b0 = ld_frag(&ldsK[nh * 16 + col][16 * half]);
      FragU b1 = ld_frag(&ldsK[nh * 16 + col][32 + 16 * half]);
      acc = __builtin_amdgcn_wmma_f32_16x16x32_bf16(
          false, qfrag[0], false, b0.v, (short)0, acc, false, false);
      acc = __builtin_amdgcn_wmma_f32_16x16x32_bf16(
          false, qfrag[1], false, b1.v, (short)0, acc, false, false);
      s[nh] = acc;
    }

    // ---- row max: step-major shuffle reduction (pipelined bpermutes) ----
    float t8[8];
    #pragma unroll
    for (int r = 0; r < 8; ++r)
      t8[r] = fmaxf(fmaxf(s[0][r], s[1][r]), fmaxf(s[2][r], s[3][r]));
    #pragma unroll
    for (int off = 8; off; off >>= 1) {
      float sh[8];
      #pragma unroll
      for (int r = 0; r < 8; ++r) sh[r] = __shfl_xor(t8[r], off, 16);
      #pragma unroll
      for (int r = 0; r < 8; ++r) t8[r] = fmaxf(t8[r], sh[r]);
    }

    // ---- rescale + exponentiate + stash P (bf16) into per-wave LDS ----
    float alpha[8];
    #pragma unroll
    for (int r = 0; r < 8; ++r) {
      float mnew = fmaxf(m_r[r], t8[r]);
      alpha[r] = __expf(m_r[r] - mnew);
      m_r[r] = mnew;
      const int M = r + 8 * half;
      #pragma unroll
      for (int nh = 0; nh < 4; ++nh) {
        float e = __expf(s[nh][r] - mnew);
        ldsP[w][M][col + 16 * nh] = f2bf(e);
      }
      o[0][r] *= alpha[r]; o[1][r] *= alpha[r];
      o[2][r] *= alpha[r]; o[3][r] *= alpha[r];
    }

    // ---- P fragments (A layout), row-sum WMMA + PV WMMAs ----
    // 5 consecutive WMMAs share A = pf.v -> reuse_a hint on leading ones
    v8f rs;
    #pragma unroll
    for (int r = 0; r < 8; ++r) rs[r] = 0.0f;
    #pragma unroll
    for (int pc = 0; pc < 2; ++pc) {
      FragU pf;
      pf.u[0] = *(const uint4*)&ldsP[w][col][pc * 32 + 16 * half];
      pf.u[1] = *(const uint4*)&ldsP[w][col][pc * 32 + 16 * half + 8];
      FragU vf0 = ld_frag(&ldsVT[0 * 16 + col][pc * 32 + 16 * half]);
      FragU vf1 = ld_frag(&ldsVT[1 * 16 + col][pc * 32 + 16 * half]);
      FragU vf2 = ld_frag(&ldsVT[2 * 16 + col][pc * 32 + 16 * half]);
      FragU vf3 = ld_frag(&ldsVT[3 * 16 + col][pc * 32 + 16 * half]);
      rs = __builtin_amdgcn_wmma_f32_16x16x32_bf16(
          false, pf.v, false, ones.v, (short)0, rs, true, false);
      o[0] = __builtin_amdgcn_wmma_f32_16x16x32_bf16(
          false, pf.v, false, vf0.v, (short)0, o[0], true, false);
      o[1] = __builtin_amdgcn_wmma_f32_16x16x32_bf16(
          false, pf.v, false, vf1.v, (short)0, o[1], true, false);
      o[2] = __builtin_amdgcn_wmma_f32_16x16x32_bf16(
          false, pf.v, false, vf2.v, (short)0, o[2], true, false);
      o[3] = __builtin_amdgcn_wmma_f32_16x16x32_bf16(
          false, pf.v, false, vf3.v, (short)0, o[3], false, false);
    }
    #pragma unroll
    for (int r = 0; r < 8; ++r) l_r[r] = l_r[r] * alpha[r] + rs[r];
  }

  // ---- epilogue: O / l, coalesced fp32 stores ----
  #pragma unroll
  for (int r = 0; r < 8; ++r) {
    float inv = 1.0f / l_r[r];
    const int mrow = m0 + r + 8 * half;
    float* op = out + base + (size_t)mrow * kD;
    #pragma unroll
    for (int t = 0; t < 4; ++t) op[t * 16 + col] = o[t][r] * inv;
  }
}

extern "C" void kernel_launch(void* const* d_in, const int* in_sizes, int n_in,
                              void* d_out, int out_size, void* d_ws, size_t ws_size,
                              hipStream_t stream) {
  (void)in_sizes; (void)n_in; (void)out_size; (void)d_ws; (void)ws_size;
  const float* q = (const float*)d_in[0];
  const float* k = (const float*)d_in[1];
  const float* v = (const float*)d_in[2];
  float* o = (float*)d_out;
  dim3 grid(kS / kBR, kB * kH);   // (16, 24)
  dim3 block(256);                // 8 waves (wave32)
  hipLaunchKernelGGL(fa_fwd_gfx1250, grid, block, 0, stream, q, k, v, o);
}
